// ReverseCrossAttention_30812095382045
// MI455X (gfx1250) — matrix operand order repaired and checked by reading, not compiled, not run on hardware
//
#include <hip/hip_runtime.h>

typedef _Float16 half_t;
typedef __attribute__((ext_vector_type(16))) _Float16 v16h;
typedef __attribute__((ext_vector_type(8)))  _Float16 v8h;
typedef __attribute__((ext_vector_type(8)))  float    v8f;
typedef __attribute__((ext_vector_type(4)))  int      v4i;

// Fixed problem dims (compile-time so strides fold into immediate offsets)
constexpr int Bc = 4;
constexpr int Nc = 2048;
constexpr int Dc = 1024;
constexpr int Mc = Bc * Nc;   // 8192

#ifndef __has_builtin
#define __has_builtin(x) 0
#endif

#if defined(__HIP_DEVICE_COMPILE__) && __has_builtin(__builtin_amdgcn_global_load_async_to_lds_b128)
#define USE_ASYNC_LDS 1
#else
#define USE_ASYNC_LDS 0
#endif

__device__ __forceinline__ v8f wmma_f16(v16h a, v16h b, v8f c) {
    // D = A(16x32 f16) * B(32x16 f16) + C(16x16 f32)
    return __builtin_amdgcn_wmma_f32_16x16x32_f16(
        false, a, false, b, (short)0, c, false, false);
}

// ---- A fragment (16x32, f16) from row-major f32 source, ld = Dc ----
// packed order per lane: K = h*16+{0..15}
__device__ __forceinline__ v16h load_a_f32(const float* __restrict__ base,
                                           int m0, int k0, int lane) {
    int r = lane & 15, h = lane >> 4;
    const float* p = base + (size_t)(m0 + r) * Dc + k0 + h * 16;
    float4 f0 = *(const float4*)(p);
    float4 f1 = *(const float4*)(p + 4);
    float4 f2 = *(const float4*)(p + 8);
    float4 f3 = *(const float4*)(p + 12);
    v16h a;
    a[0]=(half_t)f0.x; a[1]=(half_t)f0.y; a[2]=(half_t)f0.z; a[3]=(half_t)f0.w;
    a[4]=(half_t)f1.x; a[5]=(half_t)f1.y; a[6]=(half_t)f1.z; a[7]=(half_t)f1.w;
    a[8]=(half_t)f2.x; a[9]=(half_t)f2.y; a[10]=(half_t)f2.z; a[11]=(half_t)f2.w;
    a[12]=(half_t)f3.x; a[13]=(half_t)f3.y; a[14]=(half_t)f3.z; a[15]=(half_t)f3.w;
    return a;
}

// ---- A fragment (16x32, f16) from row-major f16 source (LDS or global) ----
template <int LD>
__device__ __forceinline__ v16h load_a_f16(const half_t* __restrict__ base,
                                           int m0, int k0, int lane) {
    int r = lane & 15, h = lane >> 4;
    const half_t* p = base + (size_t)(m0 + r) * LD + k0 + h * 16;
    v8h x0 = *(const v8h*)(p);
    v8h x1 = *(const v8h*)(p + 8);
    v16h a;
#pragma unroll
    for (int j = 0; j < 8; ++j) { a[j] = x0[j]; a[8 + j] = x1[j]; }
    return a;
}

// ---- B fragment (32x16) where B = src^T (Q*K^T): contiguous 32B per lane ----
__device__ __forceinline__ v16h load_bt_f16(const half_t* __restrict__ src,
                                            int n0, int k0, int lane) {
    int col = lane & 15, h = lane >> 4;
    const half_t* p = src + (size_t)(n0 + col) * Dc + k0 + h * 16;
    return *(const v16h*)p;
}

// ---- B fragment from transposed LDS tile [cols][32 k]: contiguous 32B ----
__device__ __forceinline__ v16h load_b_ldsT(const half_t* tile, int csub, int lane) {
    int col = lane & 15, h = lane >> 4;
    return *(const v16h*)&tile[(csub * 16 + col) * 32 + h * 16];
}

// ---- async (or fallback) copy of a 32 KB f16 tile global -> LDS ----
__device__ __forceinline__ void stage_qtile(const half_t* __restrict__ src,
                                            half_t* dst_lds, int tid) {
#if USE_ASYNC_LDS
#pragma unroll
    for (int i = 0; i < 8; ++i) {
        const half_t* g = src + (size_t)(i * 256 + tid) * 8;
        half_t* l = dst_lds + (size_t)(i * 256 + tid) * 8;
        __builtin_amdgcn_global_load_async_to_lds_b128(
            (__attribute__((address_space(1))) v4i*)g,
            (__attribute__((address_space(3))) v4i*)l,
            0, 0);
    }
#if __has_builtin(__builtin_amdgcn_s_wait_asynccnt)
    __builtin_amdgcn_s_wait_asynccnt(0);
#else
    asm volatile("s_wait_asynccnt 0x0" ::: "memory");
#endif
#else
    const uint4* s4 = (const uint4*)src;
    uint4* d4 = (uint4*)dst_lds;
    for (int i = tid; i < 2048; i += 256) d4[i] = s4[i];
#endif
}

// =====================================================================
// Kernel 1: fused QKV projection  Y = cvt_f16(X * W + bias)
// block tile 128x128 (8 waves, each 16x128 = 8 WMMA/k-step), double-buffered
// grid: (Dc/128, Mc/128, 3)   block: 256
// =====================================================================
__global__ __launch_bounds__(256) void qkv_gemm_kernel(
    const float* __restrict__ x1, const float* __restrict__ x2,
    const float* __restrict__ Wq, const float* __restrict__ Wk, const float* __restrict__ Wv,
    const float* __restrict__ bq, const float* __restrict__ bk, const float* __restrict__ bv,
    half_t* __restrict__ Qh, half_t* __restrict__ Kh, half_t* __restrict__ Vh)
{
    const float* X; const float* W; const float* bias; half_t* Y;
    if (blockIdx.z == 0)      { X = x1; W = Wq; bias = bq; Y = Qh; }
    else if (blockIdx.z == 1) { X = x2; W = Wk; bias = bk; Y = Kh; }
    else                      { X = x2; W = Wv; bias = bv; Y = Vh; }

    __shared__ __align__(16) half_t wt[2][128 * 32];   // transposed [col][k], 8KB each
    int tid = threadIdx.x, lane = tid & 31, wave = tid >> 5;
    int m0 = blockIdx.y * 128 + wave * 16;
    int n0 = blockIdx.x * 128;
    int sc   = tid & 127;            // staged column 0..127
    int sk16 = (tid >> 7) * 16;      // staged k-base {0,16}

    auto stage = [&](int buf, int k0) {
        const float* p = W + (size_t)(k0 + sk16) * Dc + n0 + sc;
        v8h h0, h1;
#pragma unroll
        for (int j = 0; j < 8; ++j) { h0[j] = (half_t)p[0]; p += Dc; }
#pragma unroll
        for (int j = 0; j < 8; ++j) { h1[j] = (half_t)p[0]; p += Dc; }
        *(v8h*)&wt[buf][sc * 32 + sk16]     = h0;
        *(v8h*)&wt[buf][sc * 32 + sk16 + 8] = h1;
    };

    v8f acc[8];
    v8f zero = {};
#pragma unroll
    for (int c = 0; c < 8; ++c) acc[c] = zero;

    stage(0, 0);
    __syncthreads();

    constexpr int nsteps = Dc / 32;
    for (int s = 0; s < nsteps; ++s) {
        int k0 = s * 32, cur = s & 1;
        if (s + 1 < nsteps) {
            stage(1 - cur, k0 + 32);
            __builtin_prefetch(X + (size_t)(m0 + (lane & 15)) * Dc + k0 + 64, 0, 1);
        }
        v16h a = load_a_f32(X, m0, k0, lane);
        v16h bfr[8];
#pragma unroll
        for (int c = 0; c < 8; ++c) bfr[c] = load_b_ldsT(wt[cur], c, lane);
#pragma unroll
        for (int c = 0; c < 8; ++c) acc[c] = wmma_f16(a, bfr[c], acc[c]);
        __syncthreads();
    }

    int col = lane & 15, h = lane >> 4;
#pragma unroll
    for (int c = 0; c < 8; ++c) {
        float bb = bias[n0 + c * 16 + col];
#pragma unroll
        for (int r = 0; r < 8; ++r)
            Y[(size_t)(m0 + h * 8 + r) * Dc + n0 + c * 16 + col] = (half_t)(acc[c][r] + bb);
    }
}

// =====================================================================
// Kernel 2: column sums of V (per batch)
// =====================================================================
__global__ __launch_bounds__(256) void colsum_kernel(
    const half_t* __restrict__ Vh, float* __restrict__ cs)
{
    int b = blockIdx.y;
    int d = blockIdx.x * 256 + threadIdx.x;
    const half_t* Vb = Vh + (size_t)b * Nc * Dc;
    float a = 0.0f;
    for (int n = 0; n < Nc; ++n) a += (float)Vb[(size_t)n * Dc + d];
    cs[(size_t)b * Dc + d] = a;
}

// =====================================================================
// Kernel 3: softmax stats (m, l) via WMMA scores; 4 key tiles per A-load
// grid: (Nc/16, Bc)   block: 256
// =====================================================================
__global__ __launch_bounds__(256) void attn_stats_kernel(
    const half_t* __restrict__ Qh, const half_t* __restrict__ Kh,
    float* __restrict__ mbuf, float* __restrict__ lbuf)
{
    int b = blockIdx.y;
    int q0 = blockIdx.x * 16;
    const half_t* Qb = Qh + (size_t)b * Nc * Dc;
    const half_t* Kb = Kh + (size_t)b * Nc * Dc;

    __shared__ __align__(16) half_t qtile[16 * Dc];   // 32 KB
    stage_qtile(Qb + (size_t)q0 * Dc, qtile, threadIdx.x);
    __syncthreads();

    int tid = threadIdx.x, lane = tid & 31, wave = tid >> 5;
    const float scale = rsqrtf((float)Dc);
    v8f zero = {};

    float m8[8], l8[8];
#pragma unroll
    for (int r = 0; r < 8; ++r) { m8[r] = -3.0e38f; l8[r] = 0.0f; }

    for (int g = 0; g < 4; ++g) {   // wave's 16 key tiles in groups of 4
        v8f s[4];
#pragma unroll
        for (int i = 0; i < 4; ++i) s[i] = zero;
        for (int e0 = 0; e0 < Dc; e0 += 32) {
            v16h a = load_a_f16<Dc>(qtile, 0, e0, lane);
            v16h bt[4];
#pragma unroll
            for (int i = 0; i < 4; ++i)
                bt[i] = load_bt_f16(Kb, (wave + (g * 4 + i) * 8) * 16, e0, lane);
#pragma unroll
            for (int i = 0; i < 4; ++i) s[i] = wmma_f16(a, bt[i], s[i]);
        }
#pragma unroll
        for (int i = 0; i < 4; ++i) {
#pragma unroll
            for (int r = 0; r < 8; ++r) {
                float v = s[i][r] * scale;
                float t = v;
#pragma unroll
                for (int msk = 1; msk < 16; msk <<= 1) t = fmaxf(t, __shfl_xor(t, msk, 32));
                float mn = fmaxf(m8[r], t);
                float e  = __expf(v - mn);
#pragma unroll
                for (int msk = 1; msk < 16; msk <<= 1) e += __shfl_xor(e, msk, 32);
                l8[r] = l8[r] * __expf(m8[r] - mn) + e;
                m8[r] = mn;
            }
        }
    }

    __shared__ float ml[8][16][2];
    if ((lane & 15) == 0) {
        int h = lane >> 4;
#pragma unroll
        for (int r = 0; r < 8; ++r) {
            ml[wave][h * 8 + r][0] = m8[r];
            ml[wave][h * 8 + r][1] = l8[r];
        }
    }
    __syncthreads();
    if (tid < 16) {
        float M = -3.0e38f;
        for (int w = 0; w < 8; ++w) M = fmaxf(M, ml[w][tid][0]);
        float L = 0.0f;
        for (int w = 0; w < 8; ++w) L += ml[w][tid][1] * __expf(ml[w][tid][0] - M);
        mbuf[(size_t)b * Nc + q0 + tid] = M;
        lbuf[(size_t)b * Nc + q0 + tid] = L;
    }
}

// =====================================================================
// Kernel 4: P = exp(scores*scale - m)/l (f16); 4 key tiles per wave
// grid: (Nc/16/32, Nc/16, Bc)   block: 256
// =====================================================================
__global__ __launch_bounds__(256) void pmat_kernel(
    const half_t* __restrict__ Qh, const half_t* __restrict__ Kh,
    const float* __restrict__ mbuf, const float* __restrict__ lbuf,
    half_t* __restrict__ P)
{
    int b = blockIdx.z;
    int q0 = blockIdx.y * 16;
    const half_t* Qb = Qh + (size_t)b * Nc * Dc;
    const half_t* Kb = Kh + (size_t)b * Nc * Dc;

    __shared__ __align__(16) half_t qtile[16 * Dc];   // 32 KB
    stage_qtile(Qb + (size_t)q0 * Dc, qtile, threadIdx.x);
    __syncthreads();

    int tid = threadIdx.x, lane = tid & 31, wave = tid >> 5;
    int jbase = blockIdx.x * 32 + wave * 4;
    const float scale = rsqrtf((float)Dc);
    v8f zero = {};

    v8f s[4];
#pragma unroll
    for (int i = 0; i < 4; ++i) s[i] = zero;
    for (int e0 = 0; e0 < Dc; e0 += 32) {
        v16h a = load_a_f16<Dc>(qtile, 0, e0, lane);
        v16h bt[4];
#pragma unroll
        for (int i = 0; i < 4; ++i)
            bt[i] = load_bt_f16(Kb, (jbase + i) * 16, e0, lane);
#pragma unroll
        for (int i = 0; i < 4; ++i) s[i] = wmma_f16(a, bt[i], s[i]);
    }

    int col = lane & 15, h = lane >> 4;
    half_t* Pb = P + ((size_t)b * Nc + q0) * Nc;
    float mm[8], li[8];
#pragma unroll
    for (int r = 0; r < 8; ++r) {
        int row = h * 8 + r;
        mm[r] = mbuf[(size_t)b * Nc + q0 + row];
        li[r] = 1.0f / lbuf[(size_t)b * Nc + q0 + row];
    }
#pragma unroll
    for (int i = 0; i < 4; ++i) {
#pragma unroll
        for (int r = 0; r < 8; ++r) {
            float p = __expf(s[i][r] * scale - mm[r]) * li[r];
            Pb[(size_t)(h * 8 + r) * Nc + (jbase + i) * 16 + col] = (half_t)p;
        }
    }
}

// =====================================================================
// Kernel 5: ctx = (colsum(V) - P @ V) / (N-1)   (f32)
// block tile 128x128 (8 waves x 16x128), double-buffered V tile in LDS
// grid: (Dc/128, Nc/128, Bc)   block: 256
// =====================================================================
__global__ __launch_bounds__(256) void pv_gemm_kernel(
    const half_t* __restrict__ P, const half_t* __restrict__ Vh,
    const float* __restrict__ cs, float* __restrict__ ctx, float invNm1)
{
    int b = blockIdx.z;
    const half_t* Pb = P + (size_t)b * Nc * Nc;
    const half_t* Vb = Vh + (size_t)b * Nc * Dc;

    __shared__ __align__(16) half_t vt[2][128 * 32];   // transposed [col][k], 8KB each
    int tid = threadIdx.x, lane = tid & 31, wave = tid >> 5;
    int m0 = blockIdx.y * 128 + wave * 16;
    int n0 = blockIdx.x * 128;
    int sc   = tid & 127;
    int sk16 = (tid >> 7) * 16;

    auto stage = [&](int buf, int k0) {
        const half_t* p = Vb + (size_t)(k0 + sk16) * Dc + n0 + sc;
        v8h h0, h1;
#pragma unroll
        for (int j = 0; j < 8; ++j) { h0[j] = p[0]; p += Dc; }
#pragma unroll
        for (int j = 0; j < 8; ++j) { h1[j] = p[0]; p += Dc; }
        *(v8h*)&vt[buf][sc * 32 + sk16]     = h0;
        *(v8h*)&vt[buf][sc * 32 + sk16 + 8] = h1;
    };

    v8f acc[8];
    v8f zero = {};
#pragma unroll
    for (int c = 0; c < 8; ++c) acc[c] = zero;

    stage(0, 0);
    __syncthreads();

    constexpr int nsteps = Nc / 32;
    for (int s = 0; s < nsteps; ++s) {
        int k0 = s * 32, cur = s & 1;
        if (s + 1 < nsteps) stage(1 - cur, k0 + 32);
        v16h a = load_a_f16<Nc>(Pb, m0, k0, lane);
        v16h bfr[8];
#pragma unroll
        for (int c = 0; c < 8; ++c) bfr[c] = load_b_ldsT(vt[cur], c, lane);
#pragma unroll
        for (int c = 0; c < 8; ++c) acc[c] = wmma_f16(a, bfr[c], acc[c]);
        __syncthreads();
    }

    int col = lane & 15, h = lane >> 4;
    float* cb = ctx + ((size_t)b * Nc + m0) * Dc;
#pragma unroll
    for (int c = 0; c < 8; ++c) {
        float c0 = cs[(size_t)b * Dc + n0 + c * 16 + col];
#pragma unroll
        for (int r = 0; r < 8; ++r)
            cb[(size_t)(h * 8 + r) * Dc + n0 + c * 16 + col] = (c0 - acc[c][r]) * invNm1;
    }
}

// =====================================================================
// Kernel 6: LayerNorm over last dim + ReLU (single float4 load per elem)
// grid: (Mc)   block: 256
// =====================================================================
__global__ __launch_bounds__(256) void ln_relu_kernel(
    const float* __restrict__ ctx, const float* __restrict__ gamma,
    const float* __restrict__ beta, float* __restrict__ out)
{
    size_t row = blockIdx.x;
    const float4* x4 = (const float4*)(ctx + row * Dc);
    float4* y4 = (float4*)(out + row * Dc);
    int tid = threadIdx.x, lane = tid & 31, wave = tid >> 5;

    float4 v = x4[tid];
    float s  = v.x + v.y + v.z + v.w;
    float s2 = v.x * v.x + v.y * v.y + v.z * v.z + v.w * v.w;
#pragma unroll
    for (int msk = 1; msk < 32; msk <<= 1) {
        s  += __shfl_xor(s,  msk, 32);
        s2 += __shfl_xor(s2, msk, 32);
    }
    __shared__ float red[2][8];
    __shared__ float stats[2];
    if (lane == 0) { red[0][wave] = s; red[1][wave] = s2; }
    __syncthreads();
    if (tid == 0) {
        float ts = 0.0f, t2 = 0.0f;
        for (int w = 0; w < 8; ++w) { ts += red[0][w]; t2 += red[1][w]; }
        stats[0] = ts; stats[1] = t2;
    }
    __syncthreads();
    float mu  = stats[0] / (float)Dc;
    float var = stats[1] / (float)Dc - mu * mu;
    float inv = rsqrtf(var + 1e-5f);
    float4 g = ((const float4*)gamma)[tid];
    float4 bb = ((const float4*)beta)[tid];
    float4 o;
    o.x = fmaxf((v.x - mu) * inv * g.x + bb.x, 0.0f);
    o.y = fmaxf((v.y - mu) * inv * g.y + bb.y, 0.0f);
    o.z = fmaxf((v.z - mu) * inv * g.z + bb.z, 0.0f);
    o.w = fmaxf((v.w - mu) * inv * g.w + bb.w, 0.0f);
    y4[tid] = o;
}

// =====================================================================
extern "C" void kernel_launch(void* const* d_in, const int* in_sizes, int n_in,
                              void* d_out, int out_size, void* d_ws, size_t ws_size,
                              hipStream_t stream) {
    (void)in_sizes; (void)n_in; (void)out_size; (void)ws_size;

    const float* x1    = (const float*)d_in[0];
    const float* x2    = (const float*)d_in[1];
    const float* Wq    = (const float*)d_in[2];
    const float* bq    = (const float*)d_in[3];
    const float* Wk    = (const float*)d_in[4];
    const float* bk    = (const float*)d_in[5];
    const float* Wv    = (const float*)d_in[6];
    const float* bv    = (const float*)d_in[7];
    const float* gamma = (const float*)d_in[8];
    const float* beta  = (const float*)d_in[9];
    float* out = (float*)d_out;

    char* ws = (char*)d_ws;
    size_t off = 0;
    auto alloc = [&](size_t bytes) -> void* {
        void* p = ws + off;
        off = (off + bytes + 255) & ~(size_t)255;
        return p;
    };
    half_t* Qh   = (half_t*)alloc((size_t)Mc * Dc * sizeof(half_t));
    half_t* Kh   = (half_t*)alloc((size_t)Mc * Dc * sizeof(half_t));
    half_t* Vh   = (half_t*)alloc((size_t)Mc * Dc * sizeof(half_t));
    half_t* P    = (half_t*)alloc((size_t)Bc * Nc * Nc * sizeof(half_t));
    float*  ctx  = (float*) alloc((size_t)Mc * Dc * sizeof(float));
    float*  cs   = (float*) alloc((size_t)Bc * Dc * sizeof(float));
    float*  mbuf = (float*) alloc((size_t)Mc * sizeof(float));
    float*  lbuf = (float*) alloc((size_t)Mc * sizeof(float));

    qkv_gemm_kernel<<<dim3(Dc / 128, Mc / 128, 3), 256, 0, stream>>>(
        x1, x2, Wq, Wk, Wv, bq, bk, bv, Qh, Kh, Vh);

    colsum_kernel<<<dim3(Dc / 256, Bc), 256, 0, stream>>>(Vh, cs);

    attn_stats_kernel<<<dim3(Nc / 16, Bc), 256, 0, stream>>>(Qh, Kh, mbuf, lbuf);

    pmat_kernel<<<dim3(Nc / (16 * 32), Nc / 16, Bc), 256, 0, stream>>>(
        Qh, Kh, mbuf, lbuf, P);

    pv_gemm_kernel<<<dim3(Dc / 128, Nc / 128, Bc), 256, 0, stream>>>(
        P, Vh, cs, ctx, 1.0f / (float)(Nc - 1));

    ln_relu_kernel<<<dim3(Mc), 256, 0, stream>>>(ctx, gamma, beta, out);
}
